// ModulatedConv2d_446676599370
// MI455X (gfx1250) — matrix operand order repaired and checked by reading, compiled, not run on hardware
//
#include <hip/hip_runtime.h>
#include <hip/hip_bf16.h>
#include <cstddef>

typedef __bf16 bf16_t;
typedef __attribute__((ext_vector_type(16))) __bf16 v16bf;
typedef __attribute__((ext_vector_type(8)))  __bf16 v8bf;
typedef __attribute__((ext_vector_type(8)))  float  v8f;

#define BATCH   16
#define IN_C    512
#define OUT_C   512
#define HW      64
#define KTILES  144            // 16 ic-tiles * 9 taps
#define CONV_SCALE 1.4731391274719739e-2f   // 1/sqrt(512*9)
#define MOD_SCALE  4.4194173824159216e-2f   // 1/sqrt(512)
#define EPS 1e-8f

// ---------------- kernel 1: style modulation s[b][ic] ----------------
__global__ void mod_kernel(const float* __restrict__ style,
                           const float* __restrict__ mw,
                           const float* __restrict__ mb,
                           float* __restrict__ s) {
    int idx = blockIdx.x * blockDim.x + threadIdx.x;   // 16*512
    int b = idx >> 9, i = idx & 511;
    const float* wr = mw + (size_t)i * 512;
    const float* sr = style + (size_t)b * 512;
    float acc = 0.f;
    #pragma unroll 4
    for (int k = 0; k < 512; ++k) acc += sr[k] * wr[k];
    s[idx] = acc * MOD_SCALE + mb[i];
}

// ---------------- kernel 2: per (oc,ic) sum of squared taps ----------------
__global__ void wsq_kernel(const float* __restrict__ w, float* __restrict__ wsq) {
    int idx = blockIdx.x * blockDim.x + threadIdx.x;   // 512*512
    const float* p = w + (size_t)idx * 9;
    float acc = 0.f;
    #pragma unroll
    for (int t = 0; t < 9; ++t) acc += p[t] * p[t];
    wsq[idx] = acc;
}

// ---------------- kernel 3: demod[b][oc] ----------------
__global__ void demod_kernel(const float* __restrict__ s,
                             const float* __restrict__ wsq,
                             float* __restrict__ demod) {
    int idx = blockIdx.x * blockDim.x + threadIdx.x;   // 16*512
    int b = idx >> 9, oc = idx & 511;
    const float* sr = s + (size_t)b * 512;
    const float* wr = wsq + (size_t)oc * 512;
    float acc = 0.f;
    #pragma unroll 4
    for (int ic = 0; ic < 512; ++ic) acc += sr[ic] * sr[ic] * wr[ic];
    demod[idx] = rsqrtf(CONV_SCALE * CONV_SCALE * acc + EPS);
}

// ---------------- kernel 4: pack modulated bf16 weights, A-fragment order ----
// Layout: [b][ocSub(32)][ictap(144 = ict*9+tap)][lane(32)][e(16)]
// A k order used here (contiguous, same as B gather): lane<16 -> row=lane, K={0..15};
// lane>=16 -> row=lane-16, K={16..31}.
__global__ void pack_kernel(const float* __restrict__ w,
                            const float* __restrict__ s,
                            const float* __restrict__ demod,
                            bf16_t* __restrict__ apk) {
    int t = blockIdx.x * blockDim.x + threadIdx.x;     // 16*32*144*32
    int lane = t & 31;
    int r = t >> 5;
    int ictap = r % KTILES; r /= KTILES;
    int ocSub = r & 31;
    int b = r >> 5;
    int ict = ictap / 9, tap = ictap - ict * 9;
    int row = lane & 15, hi = lane >> 4;
    int oc = ocSub * 16 + row;
    float dm = demod[b * 512 + oc] * CONV_SCALE;
    const float* sb = s + (size_t)b * 512;
    v16bf o;
    #pragma unroll
    for (int e = 0; e < 16; ++e) {
        int col = e + (hi << 4);
        int ic = ict * 32 + col;
        float v = w[((size_t)oc * 512 + ic) * 9 + tap] * sb[ic] * dm;
        o[e] = (__bf16)v;
    }
    *(v16bf*)(apk + (size_t)t * 16) = o;
}

// ---------------- kernel 5: implicit-GEMM conv via WMMA ----------------
// Block: 256 threads = 8 wave32. Tile: 64 oc x 256 px (16x16 spatial).
// Wave w: M-subtile m=w&3; nh=w>>2 selects rows nh*8..nh*8+7 (8 N-subtiles,
// one spatial row of 16 px each) -> 8 f32 accumulators per wave.
// Per ic-tile: stage 32ic x 18x18 halo (bf16, zero-padded) once; 9 taps are
// pure immediate ds offsets; 72 WMMAs between one barrier pair.
// Staging is register-double-buffered: ict+1 global loads issue before the
// 72-WMMA burst of ict, so memory latency hides behind the matrix pipe.
__global__ __launch_bounds__(256)
void conv_kernel(const float* __restrict__ input,
                 const bf16_t* __restrict__ apk,
                 float* __restrict__ out) {
    // halo: 324 pixels, 40 bf16 (32 ic + pad) per pixel = 80B rows, 16B aligned
    __shared__ __align__(16) __bf16 Bt[324 * 40];

    const int b   = blockIdx.z;
    const int ocT = blockIdx.y;                 // 0..7
    const int pT  = blockIdx.x;                 // 0..15
    const int ty = (pT >> 2) * 16, tx = (pT & 3) * 16;

    const int tid  = threadIdx.x;
    const int lane = tid & 31;
    const int wave = tid >> 5;
    const int m  = wave & 3;
    const int nh = wave >> 2;

    // ---- staging plan (invariant across ic-tiles): 1296 units of 8 ic ----
    int  stLds[6]; int stOff[6]; bool stV[6]; bool stA[6];
    #pragma unroll
    for (int u = 0; u < 6; ++u) {
        int idx = tid + u * 256;
        stA[u] = idx < 1296;
        int hp = idx >> 2, icg = idx & 3;
        int hy = hp / 18, hx = hp - hy * 18;
        int gy = ty + hy - 1, gx = tx + hx - 1;
        stV[u] = (gy >= 0) & (gy < HW) & (gx >= 0) & (gx < HW);
        stOff[u] = icg * 8 * (HW * HW) + gy * HW + gx;
        stLds[u] = hp * 80 + icg * 16;
    }

    const int ocSub = ocT * 4 + m;
    const bf16_t* abase = apk + ((size_t)(b * 32 + ocSub) * KTILES) * 512 + lane * 16;
    const float*  ibase = input + (size_t)b * IN_C * (HW * HW);

    // gather base: pixel (i = nh*8, j = lane&15), K half = (lane>>4)*16
    const int vbase = ((nh * 8) * 18 + (lane & 15)) * 80 + (lane >> 4) * 32;
    // tap (dy,dx) -> halo shift (dy+1)*18 + (dx+1) pixels, x80 bytes
    constexpr int tapOff[9] = {0, 80, 160, 1440, 1520, 1600, 2880, 2960, 3040};

    v8f pv[6];   // prefetched f32 staging data (cvt deferred to store time)
    auto stageLoad = [&](int ict_) {
        const float* src = ibase + (size_t)ict_ * 32 * (HW * HW);
        #pragma unroll
        for (int u = 0; u < 6; ++u) {
            v8f t = {};
            if (stA[u] && stV[u]) {
                const float* p = src + stOff[u];
                #pragma unroll
                for (int c = 0; c < 8; ++c) t[c] = p[(size_t)c * (HW * HW)];
            }
            pv[u] = t;
        }
    };

    v8f acc[8] = {};
    stageLoad(0);

    for (int ict = 0; ict < 16; ++ict) {
        __syncthreads();                        // prior gathers done
        #pragma unroll
        for (int u = 0; u < 6; ++u) {
            if (stA[u]) {
                v8bf o;
                #pragma unroll
                for (int c = 0; c < 8; ++c) o[c] = (__bf16)pv[u][c];
                *(v8bf*)((char*)Bt + stLds[u]) = o;
            }
        }
        __syncthreads();

        if (ict < 15) stageLoad(ict + 1);       // loads hide behind WMMA burst

        const bf16_t* afrag = abase + (size_t)ict * 9 * 512;
        #pragma unroll
        for (int tap = 0; tap < 9; ++tap) {
            v16bf a = *(const v16bf*)(afrag + tap * 512);     // 2x global b128
            #pragma unroll
            for (int q = 0; q < 8; ++q) {
                v16bf bq = *(const v16bf*)((const char*)Bt +
                              (vbase + q * 1440 + tapOff[tap])); // 2x ds b128
                acc[q] = __builtin_amdgcn_wmma_f32_16x16x32_bf16(
                              false, a, false, bq, (short)0, acc[q], false, false);
            }
        }
    }

    // ---- writeback: lane l, reg r -> M = r + (l>>4)*8, N = l&15 ----
    const int mrow = (lane >> 4) * 8;
    const int ncol = lane & 15;
    #pragma unroll
    for (int q = 0; q < 8; ++q) {
        int ypix = ty + nh * 8 + q;
        #pragma unroll
        for (int r = 0; r < 8; ++r) {
            int oc = ocT * 64 + m * 16 + mrow + r;
            out[(((size_t)b * OUT_C + oc) * HW + ypix) * HW + tx + ncol] = acc[q][r];
        }
    }
}

extern "C" void kernel_launch(void* const* d_in, const int* in_sizes, int n_in,
                              void* d_out, int out_size, void* d_ws, size_t ws_size,
                              hipStream_t stream) {
    const float* input  = (const float*)d_in[0];   // [16,512,64,64]
    const float* style  = (const float*)d_in[1];   // [16,512]
    const float* weight = (const float*)d_in[2];   // [1,512,512,3,3]
    const float* mw     = (const float*)d_in[3];   // [512,512]
    const float* mb     = (const float*)d_in[4];   // [512]
    float* out = (float*)d_out;

    float* s     = (float*)d_ws;                       // 8192
    float* demod = s + 8192;                           // 8192
    float* wsq   = demod + 8192;                       // 262144
    bf16_t* apk  = (bf16_t*)(wsq + 262144);            // 16*32*144*32*16 bf16

    mod_kernel  <<<32,   256, 0, stream>>>(style, mw, mb, s);
    wsq_kernel  <<<1024, 256, 0, stream>>>(weight, wsq);
    demod_kernel<<<32,   256, 0, stream>>>(s, wsq, demod);
    pack_kernel <<<9216, 256, 0, stream>>>(weight, s, demod, apk);

    dim3 grid(16, 8, 16);   // pixel tiles, oc tiles, batch
    conv_kernel<<<grid, 256, 0, stream>>>(input, apk, out);
}
